// HWCPEncoder_43533788512755
// MI455X (gfx1250) — matrix operand order repaired and checked by reading, not compiled, not run on hardware
//
#include <hip/hip_runtime.h>
#include <hip/hip_bf16.h>

// ---------------------------------------------------------------------------
// HWCPEncoder for MI455X (gfx1250, wave32, WMMA)
//   S=8 B=64 L=128 E=128 V=32000 K=8 M=14 H=512 3H=1536
// Core: per-timestep fused GRU step kernel using v_wmma_f32_16x16x32_bf16,
// with A-tiles (x_t rows + h rows) staged once per block into LDS via
// GLOBAL_LOAD_ASYNC_TO_LDS_B128 (ASYNCcnt) when the builtin is available.
// ---------------------------------------------------------------------------

typedef __bf16 bf16;
typedef bf16 v16bf __attribute__((ext_vector_type(16)));
typedef bf16 v8bf  __attribute__((ext_vector_type(8)));
typedef float v8f  __attribute__((ext_vector_type(8)));
typedef int  v4i   __attribute__((ext_vector_type(4)));

union V16u { v16bf v; v8bf h[2]; };

#define GH 512          // hidden
#define G3H 1536
#define GB 64           // batch
#define GL 128          // seq len
#define GE 128          // embed
#define GS 8
#define GK 8
#define GM 14

#define AS1 __attribute__((address_space(1)))
#define AS3 __attribute__((address_space(3)))

#if defined(__has_builtin)
#if __has_builtin(__builtin_amdgcn_global_load_async_to_lds_b128) && \
    __has_builtin(__builtin_amdgcn_s_wait_asynccnt)
#define USE_ASYNC_LDS 1
#endif
#endif
#ifndef USE_ASYNC_LDS
#define USE_ASYNC_LDS 0
#endif

// copy 16 bytes (8 bf16) global -> LDS
static __device__ inline void stage16(const bf16* __restrict__ g, bf16* __restrict__ l) {
#if USE_ASYNC_LDS
  __builtin_amdgcn_global_load_async_to_lds_b128((AS1 v4i*)(bf16*)g, (AS3 v4i*)l, 0, 0);
#else
  *(v8bf*)l = *(const v8bf*)g;
#endif
}

// ---- WMMA fragment loaders (layouts per CDNA5 ISA 7.12.2, bf16) -----------
// A (16x32, MxK) from LDS tile (16 rows x stride):
// lane<16 -> M=lane,    K in {kb*32+0..7, kb*32+16..23}
// lane>=16-> M=lane-16, K in {kb*32+8..15, kb*32+24..31}
static __device__ inline v16bf load_a_sh(const bf16* base, int stride, int kb, int lane) {
  const bf16* p = base + (lane & 15) * stride + kb * 32 + ((lane >> 4) << 3);
  V16u r;
  r.h[0] = *(const v8bf*)(p);
  r.h[1] = *(const v8bf*)(p + 16);
  return r.v;
}
// B (32x16, KxN) from row-major W[row][k] with row = n-col of output:
// lane<16 -> N=lane, K in {kb*32+0..7, kb*32+16..23}; lane>=16 -> N=lane-16, K in {kb*32+8..15, kb*32+24..31}
static __device__ inline v16bf load_b_bf16(const bf16* __restrict__ W,
                                           int ldk, int row0, int kb, int lane) {
  const bf16* p = W + (size_t)(row0 + (lane & 15)) * ldk
                    + kb * 32 + ((lane >> 4) << 3);
  V16u r;
  r.h[0] = *(const v8bf*)(p);
  r.h[1] = *(const v8bf*)(p + 16);
  return r.v;
}

// ---------------------------------------------------------------------------
// Fused GRU step:  gates = x_t @ Wih^T  and  h @ Whh^T   (bf16 WMMA, f32 acc)
//   r = sig(Gx_r + Gh_r + bi_r + bh_r)
//   z = sig(Gx_z + Gh_z + bi_z + bh_z)
//   n = tanh(Gx_n + bi_n + r*(Gh_n + bh_n))
//   h' = (1-z)*n + z*h
// grid = (16, 2[dir]), block = 256 (8 waves).  All 8 waves of a block share
// one 16-row batch tile -> x/h A-tiles staged once per block into LDS.
// Ping-pong h buffers between launches (no intra-step race).
// ---------------------------------------------------------------------------
__global__ void k_gru_step(const bf16* __restrict__ X,   // [L][B][D]
                           int D, int dshift, int t,
                           const bf16* __restrict__ Wih, // [2][3H][D]
                           const bf16* __restrict__ Whh, // [2][3H][H]
                           const float* __restrict__ bih,// [2][3H]
                           const float* __restrict__ bhh,// [2][3H]
                           const float* __restrict__ hIn,  // [2][B][H]
                           float* __restrict__ hOut,       // [2][B][H]
                           const bf16* __restrict__ hbIn,  // [2][B][H]
                           bf16* __restrict__ hbOut,       // [2][B][H]
                           bf16* __restrict__ y0,          // layer0: [L][B][2H]
                           float* __restrict__ outF,       // layer1 fwd [B][L][H]
                           float* __restrict__ outB)       // layer1 bwd [B][L][H]
{
  __shared__ __align__(16) bf16 sx[16 * (1024 + 8)];  // x tile, padded rows
  __shared__ __align__(16) bf16 sh[16 * (512 + 8)];   // h tile, padded rows

  const int dir  = blockIdx.y;
  const int wave = threadIdx.x >> 5;
  const int lane = threadIdx.x & 31;
  const int tid  = threadIdx.x;
  const int tri  = blockIdx.x * 8 + wave;   // 0..127
  const int ct   = tri & 31;                // H col tile (per wave)
  const int m0   = (blockIdx.x >> 2) << 4;  // batch row tile (per block)
  const int c0   = ct << 4;
  const int l    = dir ? (GL - 1 - t) : t;
  const int SXP  = D + 8;
  const int SHP  = GH + 8;

  const bf16* x   = X    + (size_t)l * GB * D;
  const bf16* wih = Wih  + (size_t)dir * G3H * D;
  const bf16* whh = Whh  + (size_t)dir * G3H * GH;
  const bf16* hb  = hbIn + (size_t)dir * GB * GH;

  // ---- stage A-tiles into LDS (16 rows of x_t, 16 rows of h) ----
  {
    const int cpr = 1 << (dshift - 3);       // 16B chunks per x row
    const int totx = 16 << (dshift - 3);
    for (int cid = tid; cid < totx; cid += 256) {
      int row = cid >> (dshift - 3);
      int col = (cid & (cpr - 1)) << 3;
      stage16(x + (size_t)(m0 + row) * D + col, sx + row * SXP + col);
    }
    for (int cid = tid; cid < 16 * 64; cid += 256) {  // 512/8 = 64 chunks per h row
      int row = cid >> 6;
      int col = (cid & 63) << 3;
      stage16(hb + (size_t)(m0 + row) * GH + col, sh + row * SHP + col);
    }
  }
  // prefetch next timestep's x tile into cache while we compute
  if (t + 1 < GL) {
    const int ln = dir ? (l - 1) : (l + 1);
    const bf16* nx = X + (size_t)ln * GB * D + (size_t)m0 * D;
    __builtin_prefetch(nx + (size_t)tid * (D >> 4), 0, 1);
  }
#if USE_ASYNC_LDS
  __builtin_amdgcn_s_wait_asynccnt(0);
#endif
  __syncthreads();

  v8f accR = {}, accZ = {}, accNx = {}, accNh = {};

  // x-projection: K = D
  const int nkx = D >> 5;
  for (int kb = 0; kb < nkx; ++kb) {
    v16bf a  = load_a_sh(sx, SXP, kb, lane);
    v16bf bR = load_b_bf16(wih, D, c0,            kb, lane);
    v16bf bZ = load_b_bf16(wih, D, GH + c0,       kb, lane);
    v16bf bN = load_b_bf16(wih, D, 2 * GH + c0,   kb, lane);
    accR  = __builtin_amdgcn_wmma_f32_16x16x32_bf16(false, a, false, bR, (short)0, accR,  false, false);
    accZ  = __builtin_amdgcn_wmma_f32_16x16x32_bf16(false, a, false, bZ, (short)0, accZ,  false, false);
    accNx = __builtin_amdgcn_wmma_f32_16x16x32_bf16(false, a, false, bN, (short)0, accNx, false, false);
  }
  // h-projection: K = H = 512
  for (int kb = 0; kb < (GH >> 5); ++kb) {
    v16bf a  = load_a_sh(sh, SHP, kb, lane);
    v16bf bR = load_b_bf16(whh, GH, c0,          kb, lane);
    v16bf bZ = load_b_bf16(whh, GH, GH + c0,     kb, lane);
    v16bf bN = load_b_bf16(whh, GH, 2 * GH + c0, kb, lane);
    accR  = __builtin_amdgcn_wmma_f32_16x16x32_bf16(false, a, false, bR, (short)0, accR,  false, false);
    accZ  = __builtin_amdgcn_wmma_f32_16x16x32_bf16(false, a, false, bZ, (short)0, accZ,  false, false);
    accNh = __builtin_amdgcn_wmma_f32_16x16x32_bf16(false, a, false, bN, (short)0, accNh, false, false);
  }

  // gate math.  C layout: VGPR r -> M = r + 8*(lane>=16), N = lane&15
  const float* bi = bih + dir * G3H;
  const float* bh = bhh + dir * G3H;
  const int c = c0 + (lane & 15);
  const float biR = bi[c] + bh[c];
  const float biZ = bi[GH + c] + bh[GH + c];
  const float biN = bi[2 * GH + c];
  const float bhN = bh[2 * GH + c];

  const float* hi = hIn  + (size_t)dir * GB * GH;
  float* ho       = hOut + (size_t)dir * GB * GH;
  bf16*  hbo      = hbOut + (size_t)dir * GB * GH;

#pragma unroll
  for (int rr = 0; rr < 8; ++rr) {
    const int m = m0 + rr + ((lane >> 4) << 3);
    const float r  = 1.f / (1.f + __expf(-(accR[rr] + biR)));
    const float zg = 1.f / (1.f + __expf(-(accZ[rr] + biZ)));
    const float n  = tanhf(accNx[rr] + biN + r * (accNh[rr] + bhN));
    const float hp = hi[(size_t)m * GH + c];
    const float hn = (1.f - zg) * n + zg * hp;
    ho[(size_t)m * GH + c]  = hn;
    hbo[(size_t)m * GH + c] = (bf16)hn;
    if (y0) {
      y0[((size_t)l * GB + m) * (2 * GH) + dir * GH + c] = (bf16)hn;
    } else {
      float* o = dir ? outB : outF;           // [B][L][H]
      o[((size_t)m * GL + l) * GH + c] = hn;
    }
  }
}

// ---------------------------------------------------------------------------
// Small helper kernels
// ---------------------------------------------------------------------------
__global__ void k_zero_f32(float* p, int n) {
  int i = blockIdx.x * 256 + threadIdx.x; if (i < n) p[i] = 0.f;
}
__global__ void k_zero_bf16(bf16* p, int n) {
  int i = blockIdx.x * 256 + threadIdx.x; if (i < n) p[i] = (bf16)0.f;
}
__global__ void k_f2bf(const float* __restrict__ s, bf16* __restrict__ d, int n) {
  int i = blockIdx.x * 256 + threadIdx.x; if (i < n) d[i] = (bf16)s[i];
}
__global__ void k_add(float* __restrict__ dst, const float* __restrict__ src, int n) {
  int i = blockIdx.x * 256 + threadIdx.x; if (i < n) dst[i] += src[i];
}
// embedding gather: X[s][l][b][e] = bf16(emb[input_var[s][b][l]][e])
__global__ void k_embed(const int* __restrict__ iv, const float* __restrict__ emb,
                        bf16* __restrict__ X) {
  int i = blockIdx.x * 256 + threadIdx.x;       // S*L*B*E = 8388608
  int e = i & 127; int r = i >> 7;
  int b = r & 63;  r >>= 6;
  int l = r & 127; int s = r >> 7;
  int tok = iv[((size_t)s * GB + b) * GL + l];
  X[i] = (bf16)emb[(size_t)tok * GE + e];
}

static __device__ inline float wave_reduce(float v) {
  for (int off = 16; off > 0; off >>= 1) v += __shfl_xor(v, off, 32);
  return v;
}

// child attention scores: s1[b][k][m] = dot(word[b,ipos[k]], word[b,cpos[k,m]])
__global__ void k_s1(const float* __restrict__ word_s, const int* __restrict__ ipos,
                     const int* __restrict__ cpos, float* __restrict__ s1) {
  int w = (blockIdx.x * blockDim.x + threadIdx.x) >> 5;  // B*K*M = 7168 waves
  int lane = threadIdx.x & 31;
  int m = w % GM; int k = (w / GM) % GK; int b = w / (GM * GK);
  const float* q  = word_s + ((size_t)b * GL + ipos[k]) * GH;
  const float* lf = word_s + ((size_t)b * GL + cpos[k * GM + m]) * GH;
  float acc = 0.f;
  for (int h = lane; h < GH; h += 32) acc += q[h] * lf[h];
  acc = wave_reduce(acc);
  if (lane == 0) s1[w] = acc;
}
__global__ void k_softmax(float* __restrict__ x, int rows, int n) {
  int r = blockIdx.x * blockDim.x + threadIdx.x;
  if (r >= rows) return;
  float* p = x + (size_t)r * n;
  float mx = -1e30f;
  for (int i = 0; i < n; ++i) mx = fmaxf(mx, p[i]);
  float s = 0.f;
  for (int i = 0; i < n; ++i) { float e = __expf(p[i] - mx); p[i] = e; s += e; }
  float inv = 1.f / s;
  for (int i = 0; i < n; ++i) p[i] *= inv;
}
__global__ void k_mix1(const float* __restrict__ word_s, const int* __restrict__ cpos,
                       const float* __restrict__ a1, float* __restrict__ mix1) {
  int i = blockIdx.x * 256 + threadIdx.x;        // (b,k,h) 262144
  int h = i & 511; int k = (i >> 9) & 7; int b = i >> 12;
  const float* a = a1 + (b * GK + k) * GM;
  float acc = 0.f;
  for (int m = 0; m < GM; ++m)
    acc += a[m] * word_s[((size_t)b * GL + cpos[k * GM + m]) * GH + h];
  mix1[i] = acc;
}
// child_vec[b,k,ho] = tanh(bias + [mix1,q1] @ W^T)
__global__ void k_childvec(const float* __restrict__ mix1, const float* __restrict__ word_s,
                           const int* __restrict__ ipos, const float* __restrict__ W,
                           const float* __restrict__ bias, float* __restrict__ cv) {
  int i = blockIdx.x * 256 + threadIdx.x;        // (b,k,ho)
  int ho = i & 511; int k = (i >> 9) & 7; int b = i >> 12;
  const float* mx = mix1 + ((size_t)(b * GK + k) << 9);
  const float* q  = word_s + ((size_t)b * GL + ipos[k]) * GH;
  const float* w  = W + (size_t)ho * (2 * GH);
  float acc = bias[ho];
  for (int h = 0; h < GH; ++h) acc += mx[h] * w[h];
  for (int h = 0; h < GH; ++h) acc += q[h] * w[GH + h];
  cv[i] = tanhf(acc);
}
__global__ void k_rootscore(const float* __restrict__ word_s, const float* __restrict__ cv,
                            float* __restrict__ sc) {
  int w = (blockIdx.x * blockDim.x + threadIdx.x) >> 5;  // (b,k) 512 waves
  int lane = threadIdx.x & 31;
  int k = w & 7; int b = w >> 3;
  const float* q = word_s + (size_t)b * GL * GH;           // l = 0
  const float* c = cv + ((size_t)(b * GK + k) << 9);
  float acc = 0.f;
  for (int h = lane; h < GH; h += 32) acc += q[h] * c[h];
  acc = wave_reduce(acc);
  if (lane == 0) sc[w] = acc;
}
__global__ void k_rootmix(const float* __restrict__ a, const float* __restrict__ cv,
                          float* __restrict__ mixR) {
  int i = blockIdx.x * 256 + threadIdx.x;        // (b,h) 32768
  int h = i & 511; int b = i >> 9;
  float acc = 0.f;
  for (int k = 0; k < GK; ++k) acc += a[b * GK + k] * cv[((size_t)(b * GK + k) << 9) + h];
  mixR[i] = acc;
}
__global__ void k_spanvec(const float* __restrict__ mixR, const float* __restrict__ word_s,
                          const float* __restrict__ W, const float* __restrict__ bias,
                          float* __restrict__ span_input, int s) {
  int i = blockIdx.x * 256 + threadIdx.x;        // (b,ho) 32768
  int ho = i & 511; int b = i >> 9;
  const float* mx = mixR + ((size_t)b << 9);
  const float* q  = word_s + (size_t)b * GL * GH;
  const float* w  = W + (size_t)ho * (2 * GH);
  float acc = bias[ho];
  for (int h = 0; h < GH; ++h) acc += mx[h] * w[h];
  for (int h = 0; h < GH; ++h) acc += q[h] * w[GH + h];
  span_input[((size_t)b * GS + s) * GH + ho] = tanhf(acc);
}
__global__ void k_xs(const float* __restrict__ span_input, float* __restrict__ xs) {
  int i = blockIdx.x * 256 + threadIdx.x;        // (b,s,h) 262144
  int h = i & 511; int s = (i >> 9) & 7;
  float di = __expf((float)(h & ~1) * (-9.210340371976184f / 512.f));
  float ang = (float)s * di;
  float pe = (h & 1) ? cosf(ang) : sinf(ang);
  xs[i] = span_input[i] + pe;
}
__global__ void k_fscore(const float* __restrict__ xs, const int* __restrict__ len,
                         float* __restrict__ sc) {
  int w = (blockIdx.x * blockDim.x + threadIdx.x) >> 5;  // (b,q,k) 4096 waves
  int lane = threadIdx.x & 31;
  int k = w & 7; int b = w >> 6;
  const float* xq = xs + ((size_t)((w >> 3)) << 9);      // (b*8+q)*512
  const float* xk = xs + ((size_t)(b * GS + k) << 9);
  float acc = 0.f;
  for (int h = lane; h < GH; h += 32) acc += xq[h] * xk[h];
  acc = wave_reduce(acc);
  if (lane == 0) sc[w] = (k >= len[b]) ? -1e30f : acc;
}
__global__ void k_fmix(const float* __restrict__ a, const float* __restrict__ xs,
                       float* __restrict__ mixF) {
  int i = blockIdx.x * 256 + threadIdx.x;        // (b,q,h)
  int h = i & 511; int q = (i >> 9) & 7; int b = i >> 12;
  float acc = 0.f;
  for (int k = 0; k < GS; ++k)
    acc += a[(b * GS + q) * GS + k] * xs[((size_t)(b * GS + k) << 9) + h];
  mixF[i] = acc;
}
__global__ void k_fout(const float* __restrict__ mixF, const float* __restrict__ xs,
                       const float* __restrict__ W, const float* __restrict__ bias,
                       const int* __restrict__ len, float* __restrict__ out_span) {
  int i = blockIdx.x * 256 + threadIdx.x;        // (b,q,ho)
  int ho = i & 511; int q = (i >> 9) & 7; int b = i >> 12;
  const float* mx = mixF + ((size_t)(b * GS + q) << 9);
  const float* xq = xs + ((size_t)(b * GS + q) << 9);
  const float* w  = W + (size_t)ho * (2 * GH);
  float acc = bias[ho];
  for (int h = 0; h < GH; ++h) acc += mx[h] * w[h];
  for (int h = 0; h < GH; ++h) acc += xq[h] * w[GH + h];
  float v = tanhf(acc);
  out_span[i] = (q < len[b]) ? v : 0.f;
}
__global__ void k_hidden(const float* __restrict__ out_span, const int* __restrict__ len,
                         float* __restrict__ out_hidden) {
  int i = blockIdx.x * 256 + threadIdx.x;        // (b,h) 32768
  int h = i & 511; int b = i >> 9;
  int q = len[b] - 1;
  out_hidden[i] = out_span[((size_t)(b * GS + q) << 9) + h];
}

// ---------------------------------------------------------------------------
extern "C" void kernel_launch(void* const* d_in, const int* in_sizes, int n_in,
                              void* d_out, int out_size, void* d_ws, size_t ws_size,
                              hipStream_t stream) {
  (void)in_sizes; (void)n_in; (void)out_size; (void)ws_size;

  const int*   input_var    = (const int*)d_in[0];
  const int*   span_length  = (const int*)d_in[2];
  const int*   internal_pos = (const int*)d_in[3];
  const int*   child_pos    = (const int*)d_in[4];
  const float* emb          = (const float*)d_in[5];
  const float* w_ih_l0      = (const float*)d_in[6];
  const float* w_hh_l0      = (const float*)d_in[7];
  const float* b_ih_l0      = (const float*)d_in[8];
  const float* b_hh_l0      = (const float*)d_in[9];
  const float* w_ih_l1      = (const float*)d_in[10];
  const float* w_hh_l1      = (const float*)d_in[11];
  const float* b_ih_l1      = (const float*)d_in[12];
  const float* b_hh_l1      = (const float*)d_in[13];
  const float* to_parent_w  = (const float*)d_in[14];
  const float* to_parent_b  = (const float*)d_in[15];
  const float* span_attn_w  = (const float*)d_in[16];
  const float* span_attn_b  = (const float*)d_in[17];

  float* out_span   = (float*)d_out;                          // (B,S,H)
  float* out_word   = out_span + (size_t)GB * GS * GH;        // (S,B,L,H)
  float* out_hidden = out_word + (size_t)GS * GB * GL * GH;   // (1,B,H)

  // ---- workspace carve (256B aligned) ----
  char* wp = (char*)d_ws;
  auto alloc = [&](size_t bytes) -> void* {
    void* r = (void*)wp; wp += (bytes + 255) & ~(size_t)255; return r;
  };
  bf16* Xbf   = (bf16*)alloc((size_t)GS * GL * GB * GE * 2);
  bf16* Wih0  = (bf16*)alloc((size_t)2 * G3H * GE * 2);
  bf16* Whh0  = (bf16*)alloc((size_t)2 * G3H * GH * 2);
  bf16* Wih1  = (bf16*)alloc((size_t)2 * G3H * (2 * GH) * 2);
  bf16* Whh1  = (bf16*)alloc((size_t)2 * G3H * GH * 2);
  bf16* y0buf = (bf16*)alloc((size_t)GL * GB * (2 * GH) * 2);
  float* tmpB = (float*)alloc((size_t)GB * GL * GH * 4);
  // ping-pong hidden state (parity x dir x B x H)
  const int HPAR = 2 * GB * GH;  // floats per parity (both dirs)
  float* h0f = (float*)alloc((size_t)2 * HPAR * 4);
  float* h1f = (float*)alloc((size_t)2 * HPAR * 4);
  bf16*  h0b = (bf16*)alloc((size_t)2 * HPAR * 2);
  bf16*  h1b = (bf16*)alloc((size_t)2 * HPAR * 2);
  float* s1buf   = (float*)alloc((size_t)GB * GK * GM * 4);
  float* mix1buf = (float*)alloc((size_t)GB * GK * GH * 4);
  float* cvbuf   = (float*)alloc((size_t)GB * GK * GH * 4);
  float* sc2buf  = (float*)alloc((size_t)GB * GK * 4);
  float* mixRbuf = (float*)alloc((size_t)GB * GH * 4);
  float* spanin  = (float*)alloc((size_t)GB * GS * GH * 4);
  float* xsbuf   = (float*)alloc((size_t)GB * GS * GH * 4);
  float* fscbuf  = (float*)alloc((size_t)GB * GS * GS * 4);
  float* mixFbuf = (float*)alloc((size_t)GB * GS * GH * 4);

  // ---- init: zero hidden states, convert weights, embed ----
  k_zero_f32<<<(2 * HPAR + 255) / 256, 256, 0, stream>>>(h0f, 2 * HPAR);
  k_zero_f32<<<(2 * HPAR + 255) / 256, 256, 0, stream>>>(h1f, 2 * HPAR);
  k_zero_bf16<<<(2 * HPAR + 255) / 256, 256, 0, stream>>>(h0b, 2 * HPAR);
  k_zero_bf16<<<(2 * HPAR + 255) / 256, 256, 0, stream>>>(h1b, 2 * HPAR);
  k_f2bf<<<(2 * G3H * GE + 255) / 256, 256, 0, stream>>>(w_ih_l0, Wih0, 2 * G3H * GE);
  k_f2bf<<<(2 * G3H * GH + 255) / 256, 256, 0, stream>>>(w_hh_l0, Whh0, 2 * G3H * GH);
  k_f2bf<<<(2 * G3H * 2 * GH + 255) / 256, 256, 0, stream>>>(w_ih_l1, Wih1, 2 * G3H * 2 * GH);
  k_f2bf<<<(2 * G3H * GH + 255) / 256, 256, 0, stream>>>(w_hh_l1, Whh1, 2 * G3H * GH);
  k_embed<<<(GS * GL * GB * GE) / 256, 256, 0, stream>>>(input_var, emb, Xbf);

  const int NWORD = GB * GL * GH;  // per-segment word elements

  for (int s = 0; s < GS; ++s) {
    const bf16* Xs = Xbf + (size_t)s * GL * GB * GE;
    float* word_s = out_word + (size_t)s * NWORD;

    // layer 0 (D = E = 128) -> y0buf
    for (int t = 0; t < GL; ++t) {
      k_gru_step<<<dim3(16, 2), 256, 0, stream>>>(
          Xs, GE, 7, t, Wih0, Whh0, b_ih_l0, b_hh_l0,
          h0f + (t & 1) * HPAR, h0f + ((t + 1) & 1) * HPAR,
          h0b + (t & 1) * HPAR, h0b + ((t + 1) & 1) * HPAR,
          y0buf, nullptr, nullptr);
    }
    // layer 1 (D = 2H = 1024) -> word (fwd direct, bwd via tmpB)
    for (int t = 0; t < GL; ++t) {
      k_gru_step<<<dim3(16, 2), 256, 0, stream>>>(
          y0buf, 2 * GH, 10, t, Wih1, Whh1, b_ih_l1, b_hh_l1,
          h1f + (t & 1) * HPAR, h1f + ((t + 1) & 1) * HPAR,
          h1b + (t & 1) * HPAR, h1b + ((t + 1) & 1) * HPAR,
          nullptr, word_s, tmpB);
    }
    k_add<<<(NWORD + 255) / 256, 256, 0, stream>>>(word_s, tmpB, NWORD);

    // child attention + span vector
    k_s1<<<(GB * GK * GM * 32) / 128, 128, 0, stream>>>(word_s, internal_pos, child_pos, s1buf);
    k_softmax<<<(GB * GK + 63) / 64, 64, 0, stream>>>(s1buf, GB * GK, GM);
    k_mix1<<<(GB * GK * GH) / 256, 256, 0, stream>>>(word_s, child_pos, s1buf, mix1buf);
    k_childvec<<<(GB * GK * GH) / 256, 256, 0, stream>>>(mix1buf, word_s, internal_pos,
                                                         to_parent_w, to_parent_b, cvbuf);
    k_rootscore<<<(GB * GK * 32) / 128, 128, 0, stream>>>(word_s, cvbuf, sc2buf);
    k_softmax<<<1, 64, 0, stream>>>(sc2buf, GB, GK);
    k_rootmix<<<(GB * GH) / 256, 256, 0, stream>>>(sc2buf, cvbuf, mixRbuf);
    k_spanvec<<<(GB * GH) / 256, 256, 0, stream>>>(mixRbuf, word_s, to_parent_w,
                                                   to_parent_b, spanin, s);
  }

  // final span self-attention
  k_xs<<<(GB * GS * GH) / 256, 256, 0, stream>>>(spanin, xsbuf);
  k_fscore<<<(GB * GS * GS * 32) / 128, 128, 0, stream>>>(xsbuf, span_length, fscbuf);
  k_softmax<<<(GB * GS + 63) / 64, 64, 0, stream>>>(fscbuf, GB * GS, GS);
  k_fmix<<<(GB * GS * GH) / 256, 256, 0, stream>>>(fscbuf, xsbuf, mixFbuf);
  k_fout<<<(GB * GS * GH) / 256, 256, 0, stream>>>(mixFbuf, xsbuf, span_attn_w,
                                                   span_attn_b, span_length, out_span);
  k_hidden<<<(GB * GH) / 256, 256, 0, stream>>>(out_span, span_length, out_hidden);
}
